// Attention_42417097015520
// MI455X (gfx1250) — matrix operand order repaired and checked by reading, not compiled, not run on hardware
//
#include <hip/hip_runtime.h>
#include <hip/hip_bf16.h>

// ---------------------------------------------------------------------------
// Multi-head attention (B=4, N=2048, D=768, H=12, hd=64) for gfx1250 (CDNA5).
// fp32 in -> f16 WMMA (v_wmma_f32_16x16x32_f16) -> fp32 out.
// Stages: weight transpose/convert, QKV GEMM, flash attention, proj GEMM.
// Tile copies use GLOBAL_LOAD_ASYNC_TO_LDS (ASYNCcnt) when the builtin is
// available, falling back to register-staged copies otherwise.
// ---------------------------------------------------------------------------

#define ATT_B   4
#define ATT_N   2048
#define ATT_D   768
#define ATT_H   12
#define ATT_HD  64
#define ATT_SCALE 0.125f   // 64^-0.5

typedef __attribute__((ext_vector_type(16))) _Float16 v16h;
typedef __attribute__((ext_vector_type(8)))  _Float16 v8h;
typedef __attribute__((ext_vector_type(8)))  float    v8f;
typedef __attribute__((ext_vector_type(4)))  int      v4i;

#if defined(__has_builtin)
#if __has_builtin(__builtin_amdgcn_global_load_async_to_lds_b128)
#define HAVE_ASYNC_LDS 1
#endif
#endif
#ifndef HAVE_ASYNC_LDS
#define HAVE_ASYNC_LDS 0
#endif

// 16-byte global -> LDS copy. Async path writes LDS directly (no VGPR
// round-trip, tracked by ASYNCcnt); fallback stages through registers.
__device__ __forceinline__ void copy16_g2l(const _Float16* gsrc,
                                           _Float16* ldst) {
#if HAVE_ASYNC_LDS
  __builtin_amdgcn_global_load_async_to_lds_b128((v4i*)gsrc, (v4i*)ldst, 0, 0);
#else
  *(v8h*)ldst = *(const v8h*)gsrc;
#endif
}

__device__ __forceinline__ void copy_wait() {
#if HAVE_ASYNC_LDS
  asm volatile("s_wait_asynccnt 0x0" ::: "memory");
#endif
}

// A-matrix fragment (16x32 f16), ISA 7.12.2 layout:
// lane L (l16 = L%16, half = L/16) holds row M = l16;
// elements 0..7  = K in [half*8,      half*8+8)
// elements 8..15 = K in [16 + half*8, 16 + half*8 + 8)
__device__ __forceinline__ v16h load_a_frag(const _Float16* row, int half) {
  v8h lo = *(const v8h*)(row + half * 8);
  v8h hi = *(const v8h*)(row + 16 + half * 8);
  return __builtin_shufflevector(lo, hi, 0, 1, 2, 3, 4, 5, 6, 7,
                                 8, 9, 10, 11, 12, 13, 14, 15);
}

// B-matrix fragment (32x16 f16) from an [N][K] (K-contiguous) LDS tile:
// lane L holds column N = L%16; elements 0..7 = K in [half*8, half*8+8),
// elements 8..15 = K in [16 + half*8, 16 + half*8 + 8).
__device__ __forceinline__ v16h load_b_frag(const _Float16* row, int half) {
  v8h lo = *(const v8h*)(row + half * 8);
  v8h hi = *(const v8h*)(row + 16 + half * 8);
  return __builtin_shufflevector(lo, hi, 0, 1, 2, 3, 4, 5, 6, 7,
                                 8, 9, 10, 11, 12, 13, 14, 15);
}

__device__ __forceinline__ v8f wmma16x16x32(v16h a, v16h b, v8f c) {
  return __builtin_amdgcn_wmma_f32_16x16x32_f16(
      /*neg_a=*/false, a, /*neg_b=*/false, b,
      /*c_mod=*/(short)0, c, /*reuse_a=*/false, /*reuse_b=*/false);
}

// --------------------------- weight convert --------------------------------
// W is K x Ncols row-major fp32; Wt is Ncols x K row-major f16 (transposed).
__global__ void transpose_to_f16(const float* __restrict__ W,
                                 _Float16* __restrict__ Wt,
                                 int K, int Ncols) {
  int idx = blockIdx.x * blockDim.x + threadIdx.x;
  if (idx >= K * Ncols) return;
  int n = idx / K;
  int k = idx - n * K;
  Wt[idx] = (_Float16)W[(size_t)k * Ncols + n];
}

// ------------------------------ QKV GEMM -----------------------------------
// C[8192 x 2304] = x[8192 x 768] @ Wqkv ; scatter into Q (scaled), K, V
// laid out [B, H, N, 64] f16.  Block tile 128x128, wave tile 64x32.
__global__ __launch_bounds__(256) void qkv_gemm(
    const float* __restrict__ x, const _Float16* __restrict__ Wt,
    _Float16* __restrict__ Qh, _Float16* __restrict__ Kh,
    _Float16* __restrict__ Vh) {
  __shared__ alignas(32) _Float16 As[128][32];
  __shared__ alignas(32) _Float16 Bs[128][32];   // [n][k]

  const int tid = threadIdx.x;
  const int lane = tid & 31, wid = tid >> 5;
  const int half = lane >> 4, l16 = lane & 15;
  const int m0 = blockIdx.x * 128;          // row in flattened B*N
  const int n0 = blockIdx.y * 128;          // col in 2304
  const int wm = wid >> 2, wn = wid & 3;    // 2 x 4 wave grid

  const int arow = tid >> 1, acol = (tid & 1) * 16;

  v8f acc[4][2] = {};

  for (int kk = 0; kk < ATT_D; kk += 32) {
    __syncthreads();
    {  // B tile first (async): pre-transposed f16 weights
      const _Float16* src = Wt + (size_t)(n0 + arow) * ATT_D + kk + acol;
      copy16_g2l(src, &Bs[arow][acol]);
      copy16_g2l(src + 8, &Bs[arow][acol + 8]);
    }
    {  // A tile: fp32 -> f16 convert on load (register path)
      const float* src = x + (size_t)(m0 + arow) * ATT_D + kk + acol;
      #pragma unroll
      for (int j = 0; j < 16; ++j) As[arow][acol + j] = (_Float16)src[j];
    }
    if (kk + 32 < ATT_D) {
      __builtin_prefetch(x + (size_t)(m0 + arow) * ATT_D + kk + 32 + acol, 0, 3);
      __builtin_prefetch(Wt + (size_t)(n0 + arow) * ATT_D + kk + 32 + acol, 0, 3);
    }
    copy_wait();
    __syncthreads();

    v16h b0 = load_b_frag(&Bs[wn * 32 + l16][0], half);
    v16h b1 = load_b_frag(&Bs[wn * 32 + 16 + l16][0], half);
    #pragma unroll
    for (int mi = 0; mi < 4; ++mi) {
      v16h a = load_a_frag(&As[wm * 64 + mi * 16 + l16][0], half);
      acc[mi][0] = wmma16x16x32(a, b0, acc[mi][0]);
      acc[mi][1] = wmma16x16x32(a, b1, acc[mi][1]);
    }
  }

  // Epilogue: scatter into Q/K/V [B,H,N,64]; fold softmax scale into Q.
  #pragma unroll
  for (int mi = 0; mi < 4; ++mi) {
    #pragma unroll
    for (int ni = 0; ni < 2; ++ni) {
      const int gc = n0 + wn * 32 + ni * 16 + l16;
      const int which = gc / ATT_D;          // 0=Q 1=K 2=V
      const int d = gc - which * ATT_D;
      const int head = d >> 6, ch = d & 63;
      _Float16* dst = (which == 0) ? Qh : (which == 1) ? Kh : Vh;
      const float scl = (which == 0) ? ATT_SCALE : 1.0f;
      #pragma unroll
      for (int r = 0; r < 8; ++r) {
        const int gm = m0 + wm * 64 + mi * 16 + half * 8 + r;
        const int bb = gm >> 11, nrow = gm & 2047;
        dst[(((size_t)(bb * ATT_H + head) * ATT_N + nrow) << 6) + ch] =
            (_Float16)(acc[mi][ni][r] * scl);
      }
    }
  }
}

// --------------------------- flash attention -------------------------------
// One block = one (b,h) and 128 query rows; each of 8 waves owns 16 rows.
// Key loop in tiles of 32; online softmax; scores never touch HBM.
__global__ __launch_bounds__(256) void flash_attn(
    const _Float16* __restrict__ Qh, const _Float16* __restrict__ Kh,
    const _Float16* __restrict__ Vh, _Float16* __restrict__ attn) {
  __shared__ alignas(32) _Float16 Ks[32][64];       // [key][ch]  (B-layout for QK^T)
  __shared__ alignas(32) _Float16 Vs[64][32];       // [ch][key]  (B-layout for P V)
  __shared__ alignas(32) _Float16 Ps[8][16][32];    // per-wave C->A relayout

  const int tid = threadIdx.x;
  const int lane = tid & 31, wid = tid >> 5;
  const int half = lane >> 4, l16 = lane & 15;
  const int q0 = blockIdx.x * 128;
  const int bh = blockIdx.y;                        // 0..B*H-1
  const size_t base = (size_t)bh * ATT_N * ATT_HD;

  // Q fragments for this wave's 16 rows (hd=64 => two K=32 A-fragments)
  const _Float16* qrow = Qh + base + (size_t)(q0 + wid * 16 + l16) * ATT_HD;
  const v16h qa0 = load_a_frag(qrow, half);
  const v16h qa1 = load_a_frag(qrow + 32, half);

  float mrow[8], lrow[8];
  v8f o[4] = {};
  #pragma unroll
  for (int r = 0; r < 8; ++r) { mrow[r] = -1e30f; lrow[r] = 0.0f; }

  const int ldidx = tid * 8;
  const int ldkey = ldidx >> 6, ldch = ldidx & 63;

  for (int kt = 0; kt < ATT_N; kt += 32) {
    __syncthreads();   // previous tile's WMMA reads of Ks/Vs/Ps are done
    {  // cooperative K/V tile load (32 keys x 64 ch), 8 f16 per thread
      const _Float16* ksrc = Kh + base + (size_t)(kt + ldkey) * ATT_HD + ldch;
      copy16_g2l(ksrc, &Ks[ldkey][ldch]);          // K tile: direct async copy
      const _Float16* vsrc = Vh + base + (size_t)(kt + ldkey) * ATT_HD + ldch;
      v8h vv = *(const v8h*)vsrc;                  // V tile: transpose via regs
      #pragma unroll
      for (int j = 0; j < 8; ++j) Vs[ldch + j][ldkey] = vv[j];
      if (kt + 32 < ATT_N) {
        __builtin_prefetch(ksrc + 32 * ATT_HD, 0, 3);
        __builtin_prefetch(vsrc + 32 * ATT_HD, 0, 3);
      }
    }
    copy_wait();
    __syncthreads();

    // S (16 rows x 32 keys) = Q . K^T  (scale pre-folded into Q)
    v8f s0 = {}, s1 = {};
    s0 = wmma16x16x32(qa0, load_b_frag(&Ks[l16][0], half), s0);
    s0 = wmma16x16x32(qa1, load_b_frag(&Ks[l16][32], half), s0);
    s1 = wmma16x16x32(qa0, load_b_frag(&Ks[16 + l16][0], half), s1);
    s1 = wmma16x16x32(qa1, load_b_frag(&Ks[16 + l16][32], half), s1);

    // Online softmax. C-fragment: element r lives at row (half*8+r), so the
    // 16 lanes of each half-wave hold all 32 key columns of the same 8 rows;
    // xor-shuffles with masks 1..8 reduce within each 16-lane group.
    float corr[8];
    #pragma unroll
    for (int r = 0; r < 8; ++r) {
      float mx = fmaxf(s0[r], s1[r]);
      #pragma unroll
      for (int msk = 1; msk <= 8; msk <<= 1) mx = fmaxf(mx, __shfl_xor(mx, msk));
      const float mnew = fmaxf(mrow[r], mx);
      corr[r] = __expf(mrow[r] - mnew);
      mrow[r] = mnew;
      const float p0 = __expf(s0[r] - mnew);
      const float p1 = __expf(s1[r] - mnew);
      s0[r] = p0; s1[r] = p1;
      float ls = p0 + p1;
      #pragma unroll
      for (int msk = 1; msk <= 8; msk <<= 1) ls += __shfl_xor(ls, msk);
      lrow[r] = lrow[r] * corr[r] + ls;
    }
    #pragma unroll
    for (int c = 0; c < 4; ++c)
      #pragma unroll
      for (int r = 0; r < 8; ++r) o[c][r] *= corr[r];

    // Stage P through LDS to relayout C-fragment -> A-fragment.
    #pragma unroll
    for (int r = 0; r < 8; ++r) {
      const int M = half * 8 + r;
      Ps[wid][M][l16]      = (_Float16)s0[r];
      Ps[wid][M][16 + l16] = (_Float16)s1[r];
    }
    __syncthreads();

    const v16h pa = load_a_frag(&Ps[wid][l16][0], half);
    #pragma unroll
    for (int c = 0; c < 4; ++c)
      o[c] = wmma16x16x32(pa, load_b_frag(&Vs[c * 16 + l16][0], half), o[c]);
  }

  // Normalize and store to attn buffer [B, N, D] (f16) for the proj GEMM.
  const int bidx = bh / ATT_H;
  const int hcol = (bh - bidx * ATT_H) * ATT_HD;
  #pragma unroll
  for (int c = 0; c < 4; ++c) {
    #pragma unroll
    for (int r = 0; r < 8; ++r) {
      const int row = q0 + wid * 16 + half * 8 + r;
      const float v = o[c][r] / lrow[r];
      attn[(size_t)(bidx * ATT_N + row) * ATT_D + hcol + c * 16 + l16] =
          (_Float16)v;
    }
  }
}

// ------------------------------ proj GEMM ----------------------------------
// out[8192 x 768] = attn @ Wproj + b_proj  (fp32 output)
__global__ __launch_bounds__(256) void proj_gemm(
    const _Float16* __restrict__ A, const _Float16* __restrict__ Wt,
    const float* __restrict__ bias, float* __restrict__ out) {
  __shared__ alignas(32) _Float16 As[128][32];
  __shared__ alignas(32) _Float16 Bs[128][32];

  const int tid = threadIdx.x;
  const int lane = tid & 31, wid = tid >> 5;
  const int half = lane >> 4, l16 = lane & 15;
  const int m0 = blockIdx.x * 128;
  const int n0 = blockIdx.y * 128;
  const int wm = wid >> 2, wn = wid & 3;

  const int arow = tid >> 1, acol = (tid & 1) * 16;

  v8f acc[4][2] = {};

  for (int kk = 0; kk < ATT_D; kk += 32) {
    __syncthreads();
    {
      const _Float16* asrc = A + (size_t)(m0 + arow) * ATT_D + kk + acol;
      copy16_g2l(asrc, &As[arow][acol]);
      copy16_g2l(asrc + 8, &As[arow][acol + 8]);
      const _Float16* bsrc = Wt + (size_t)(n0 + arow) * ATT_D + kk + acol;
      copy16_g2l(bsrc, &Bs[arow][acol]);
      copy16_g2l(bsrc + 8, &Bs[arow][acol + 8]);
      if (kk + 32 < ATT_D) {
        __builtin_prefetch(asrc + 32, 0, 3);
        __builtin_prefetch(bsrc + 32, 0, 3);
      }
    }
    copy_wait();
    __syncthreads();

    v16h b0 = load_b_frag(&Bs[wn * 32 + l16][0], half);
    v16h b1 = load_b_frag(&Bs[wn * 32 + 16 + l16][0], half);
    #pragma unroll
    for (int mi = 0; mi < 4; ++mi) {
      v16h a = load_a_frag(&As[wm * 64 + mi * 16 + l16][0], half);
      acc[mi][0] = wmma16x16x32(a, b0, acc[mi][0]);
      acc[mi][1] = wmma16x16x32(a, b1, acc[mi][1]);
    }
  }

  #pragma unroll
  for (int mi = 0; mi < 4; ++mi) {
    #pragma unroll
    for (int ni = 0; ni < 2; ++ni) {
      const int gc = n0 + wn * 32 + ni * 16 + l16;
      const float bv = bias[gc];
      #pragma unroll
      for (int r = 0; r < 8; ++r) {
        const int gm = m0 + wm * 64 + mi * 16 + half * 8 + r;
        out[(size_t)gm * ATT_D + gc] = acc[mi][ni][r] + bv;
      }
    }
  }
}

// ------------------------------ launcher -----------------------------------
extern "C" void kernel_launch(void* const* d_in, const int* in_sizes, int n_in,
                              void* d_out, int out_size, void* d_ws,
                              size_t ws_size, hipStream_t stream) {
  const float* x     = (const float*)d_in[0];   // [4,2048,768]
  const float* Wqkv  = (const float*)d_in[1];   // [768,2304]
  const float* Wproj = (const float*)d_in[2];   // [768,768]
  const float* bproj = (const float*)d_in[3];   // [768]
  float* out = (float*)d_out;                   // [4,2048,768]

  char* ws = (char*)d_ws;
  _Float16* Wt_qkv  = (_Float16*)ws;                              // 2304*768
  _Float16* Wt_proj = (_Float16*)(ws + 3538944);                  // 768*768
  _Float16* Qh      = (_Float16*)(ws + 3538944 + 1179648);        // [B,H,N,64]
  _Float16* Kh      = Qh + (size_t)ATT_B * ATT_H * ATT_N * ATT_HD;
  _Float16* Vh      = Kh + (size_t)ATT_B * ATT_H * ATT_N * ATT_HD;
  _Float16* attn    = Vh + (size_t)ATT_B * ATT_H * ATT_N * ATT_HD;  // [B,N,D]

  transpose_to_f16<<<(2304 * 768 + 255) / 256, 256, 0, stream>>>(
      Wqkv, Wt_qkv, ATT_D, 3 * ATT_D);
  transpose_to_f16<<<(768 * 768 + 255) / 256, 256, 0, stream>>>(
      Wproj, Wt_proj, ATT_D, ATT_D);

  dim3 g1(64, 18);
  qkv_gemm<<<g1, 256, 0, stream>>>(x, Wt_qkv, Qh, Kh, Vh);

  dim3 g2(ATT_N / 128, ATT_B * ATT_H);
  flash_attn<<<g2, 256, 0, stream>>>(Qh, Kh, Vh, attn);

  dim3 g3(64, 6);
  proj_gemm<<<g3, 256, 0, stream>>>(attn, Wt_proj, bproj, out);
}
